// Net_25890062860969
// MI455X (gfx1250) — compile-verified, matrix operand order checked by reading, not run on hardware
//
#include <hip/hip_runtime.h>
#include <hip/hip_bf16.h>
#include <math.h>

#define NN     100000
#define NN2    100032            // NN padded to 64-row tiles
#define NEDGE  1600000
#define EF     100
#define DD     200
#define LDH    224               // h/agg leading dim (K=200 padded to 32)
#define LDF    320               // feat leading dim (K=300 padded to 32)
#define NLAYER 6
#define NG     128

typedef __attribute__((ext_vector_type(16))) _Float16 v16h;
typedef __attribute__((ext_vector_type(8)))  float    v8f;

// ------------------------------------------------------------------
// Pre-pack a weight matrix (f32) into f16 WMMA B-fragment order:
//   Bp[kt][ntile][lane][e], 32 lanes x 16 halfs per 32x16 K-x-N tile.
//   k(e,lane) = (lane/16)*8 + (e/8)*16 + (e%8);  col = ntile*16 + lane%16
// Zero-fills K/N padding so the GEMM needs no bounds checks on B.
// ------------------------------------------------------------------
__global__ __launch_bounds__(256) void pack_B_kernel(
    const float* __restrict__ B, _Float16* __restrict__ Bp,
    int K, int Nc, int ldb, int transB, int KT, int NT)
{
  long long idx = (long long)blockIdx.x * 256 + threadIdx.x;
  long long total = (long long)KT * NT * 32 * 16;
  if (idx >= total) return;
  int e    = (int)(idx & 15);
  int lane = (int)((idx >> 4) & 31);
  int tile = (int)(idx >> 9);
  int nt   = tile % NT;
  int kt   = tile / NT;
  int col  = nt * 16 + (lane & 15);
  int k    = kt * 32 + ((lane >> 4) << 3) + ((e >> 3) << 4) + (e & 7);
  float v = 0.f;
  if (k < K && col < Nc)
    v = transB ? B[(size_t)col * ldb + k] : B[(size_t)k * ldb + col];
  Bp[idx] = (_Float16)v;
}

// ------------------------------------------------------------------
// WMMA GEMM:  C[M,Nc] = act( A[M x Kp] * Bp + bias )
// A: f32, padded rows (>= grid*64) and zero-padded K columns (lda=Kp).
// Bp: pre-packed f16 fragments (see pack_B_kernel), NT 16-col tiles.
// Block: 256 thr = 8 waves; tile 64(M) x 64(N); wave = 16x32 strip.
// ------------------------------------------------------------------
template<int ACT>
__global__ __launch_bounds__(256) void wmma_gemm_kernel(
    const float* __restrict__ A, const _Float16* __restrict__ Bp,
    const float* __restrict__ bias, float* __restrict__ C,
    int M, int Kp, int Nc, int lda, int ldc, int NT)
{
  // A tile staged directly in fragment order: As[warp_m][lane][e]
  __shared__ __attribute__((aligned(32))) _Float16 As[4][32][16];

  const int tid    = threadIdx.x;
  const int lane   = tid & 31;
  const int wid    = tid >> 5;
  const int warp_m = wid & 3;        // 0..3  (16-row strip)
  const int warp_n = wid >> 2;       // 0..1  (32-col strip)
  const int m0     = blockIdx.x * 64;
  const int n0     = blockIdx.y * 64;
  const int KT     = Kp >> 5;
  const int ntile0 = (n0 >> 4) + warp_n * 2;
  const v16h* __restrict__ Bf = reinterpret_cast<const v16h*>(Bp);

  v8f acc0 = {}, acc1 = {};

  for (int kt = 0; kt < KT; ++kt) {
    const int kk = kt << 5;
    // Stage A 64x32 tile (f32->f16), 4 x float2 per thread, unconditional
#pragma unroll
    for (int i = 0; i < 4; ++i) {
      int idx = i * 256 + tid;       // 0..1023
      int r   = idx >> 4;            // 0..63
      int k0  = (idx & 15) << 1;     // even k in 0..30
      const float2 v = *reinterpret_cast<const float2*>(
          A + (size_t)(m0 + r) * lda + kk + k0);
      int wm = r >> 4, lr = r & 15;
      int hf = (k0 >> 3) & 1;
      int e0 = (k0 & 7) | ((k0 >> 4) << 3);
      union { _Float16 h[2]; unsigned u; } p;
      p.h[0] = (_Float16)v.x;
      p.h[1] = (_Float16)v.y;
      *reinterpret_cast<unsigned*>(&As[wm][hf * 16 + lr][e0]) = p.u;
    }
    __syncthreads();
    // Operands: 2x ds_load_b128 for A, 4x global_load_b128 for B
    v16h a  = *reinterpret_cast<const v16h*>(&As[warp_m][lane][0]);
    v16h b0 = Bf[((size_t)kt * NT + ntile0) * 32 + lane];
    v16h b1 = Bf[((size_t)kt * NT + ntile0 + 1) * 32 + lane];
    acc0 = __builtin_amdgcn_wmma_f32_16x16x32_f16(
        false, a, false, b0, (short)0, acc0, false, false);
    acc1 = __builtin_amdgcn_wmma_f32_16x16x32_f16(
        false, a, false, b1, (short)0, acc1, false, false);
    __syncthreads();
  }

  // C/D layout: VGPR r -> M = (lane/16)*8 + r, N = lane%16
  const int lr  = lane & 15;
  const int hf  = lane >> 4;
  const int rb  = m0 + warp_m * 16 + hf * 8;
  const int c0  = n0 + warp_n * 32 + lr;
  const int c1  = c0 + 16;
#pragma unroll
  for (int r = 0; r < 8; ++r) {
    int row = rb + r;
    if (row < M) {
      if (c0 < Nc) {
        float v = acc0[r];
        if (bias) v += bias[c0];
        if (ACT == 1) v = fmaxf(v, 0.f);
        C[(size_t)row * ldc + c0] = v;
      }
      if (c1 < Nc) {
        float v = acc1[r];
        if (bias) v += bias[c1];
        if (ACT == 1) v = fmaxf(v, 0.f);
        C[(size_t)row * ldc + c1] = v;
      }
    }
  }
}

// ------------------------------------------------------------------
// h0 = pad(x) into [NN2, LDH] (zero K-pad columns and M-pad rows)
// ------------------------------------------------------------------
__global__ __launch_bounds__(256) void init_h_kernel(
    const float* __restrict__ x, float* __restrict__ h)
{
  long long i = (long long)blockIdx.x * 256 + threadIdx.x;
  if (i >= (long long)NN2 * LDH) return;
  int node = (int)(i / LDH), c = (int)(i % LDH);
  h[i] = (node < NN && c < EF) ? x[(size_t)node * EF + c] : 0.f;
}

// ------------------------------------------------------------------
// agg[dst] += t[src]  — one thread per (edge, 4-channel chunk)
// t: [NN, DD] tight;  agg: [NN2, LDH] padded
// ------------------------------------------------------------------
__global__ __launch_bounds__(256) void edge_kernel(
    const float* __restrict__ t, const long long* __restrict__ ei,
    float* __restrict__ agg)
{
  const int CH = DD / 4;  // 50 float4 chunks per edge
  long long idx = (long long)blockIdx.x * 256 + threadIdx.x;
  long long e = idx / CH;
  if (e >= NEDGE) return;
  int c = (int)(idx % CH) * 4;
  long long s = ei[e];
  long long d = ei[(long long)NEDGE + e];
  const float4 v = *reinterpret_cast<const float4*>(t + s * DD + c);
  float* p = agg + d * LDH + c;
  atomicAdd(p + 0, v.x);
  atomicAdd(p + 1, v.y);
  atomicAdd(p + 2, v.z);
  atomicAdd(p + 3, v.w);
}

// ------------------------------------------------------------------
// GRU gate math; gi/gh hold GEMM results with biases folded in
// ------------------------------------------------------------------
__global__ __launch_bounds__(256) void gru_kernel(
    const float* __restrict__ gi, const float* __restrict__ gh,
    float* __restrict__ h)
{
  long long i = (long long)blockIdx.x * 256 + threadIdx.x;
  if (i >= (long long)NN * DD) return;
  int node = (int)(i / DD), c = (int)(i % DD);
  size_t base = (size_t)node * (3 * DD);
  float i_r = gi[base + c], i_z = gi[base + DD + c], i_n = gi[base + 2 * DD + c];
  float h_r = gh[base + c], h_z = gh[base + DD + c], h_n = gh[base + 2 * DD + c];
  float r  = 1.f / (1.f + expf(-(i_r + h_r)));
  float z  = 1.f / (1.f + expf(-(i_z + h_z)));
  float nn = tanhf(i_n + r * h_n);
  size_t hi = (size_t)node * LDH + c;
  h[hi] = (1.f - z) * nn + z * h[hi];
}

// ------------------------------------------------------------------
// feat = concat(h, x) into [NN2, LDF] (zero-padded)
// ------------------------------------------------------------------
__global__ __launch_bounds__(256) void concat_kernel(
    const float* __restrict__ h, const float* __restrict__ x,
    float* __restrict__ feat)
{
  long long i = (long long)blockIdx.x * 256 + threadIdx.x;
  if (i >= (long long)NN2 * LDF) return;
  int node = (int)(i / LDF), c = (int)(i % LDF);
  float v = 0.f;
  if (node < NN) {
    if (c < DD)            v = h[(size_t)node * LDH + c];
    else if (c < DD + EF)  v = x[(size_t)node * EF + (c - DD)];
  }
  feat[i] = v;
}

// ------------------------------------------------------------------
// pooled[batch[n]] += feat3[n]  (G x 64, atomic segment-sum)
// ------------------------------------------------------------------
__global__ __launch_bounds__(256) void pool_kernel(
    const float* __restrict__ feat3, const long long* __restrict__ batch,
    float* __restrict__ pooled)
{
  long long i = (long long)blockIdx.x * 256 + threadIdx.x;
  if (i >= (long long)NN * 64) return;
  int node = (int)(i >> 6), c = (int)(i & 63);
  long long g = batch[node];
  atomicAdd(&pooled[g * 64 + c], feat3[(size_t)node * 64 + c]);
}

// ------------------------------------------------------------------
// out[g] = sigmoid(dot(pooled[g], out_W) + out_b)
// ------------------------------------------------------------------
__global__ __launch_bounds__(64) void final_kernel(
    const float* __restrict__ pooled, const float* __restrict__ W,
    const float* __restrict__ b, float* __restrict__ out)
{
  __shared__ float red[64];
  int g = blockIdx.x, t = threadIdx.x;
  red[t] = pooled[(size_t)g * 64 + t] * W[t];
  __syncthreads();
#pragma unroll
  for (int s = 32; s > 0; s >>= 1) {
    if (t < s) red[t] += red[t + s];
    __syncthreads();
  }
  if (t == 0) out[g] = 1.f / (1.f + expf(-(red[0] + b[0])));
}

// ------------------------------------------------------------------
extern "C" void kernel_launch(void* const* d_in, const int* in_sizes, int n_in,
                              void* d_out, int out_size, void* d_ws, size_t ws_size,
                              hipStream_t stream) {
  const float*     x      = (const float*)d_in[0];
  const long long* eidx   = (const long long*)d_in[1];   // int64 [2,NE]
  const long long* batch  = (const long long*)d_in[2];   // int64 [N]
  const float*     ggc_w  = (const float*)d_in[3];       // [L,D,D]
  const float*     W_ih   = (const float*)d_in[4];       // [3D,D]
  const float*     W_hh   = (const float*)d_in[5];       // [3D,D]
  const float*     b_ih   = (const float*)d_in[6];       // [3D]
  const float*     b_hh   = (const float*)d_in[7];       // [3D]
  const float*     mlp1_W = (const float*)d_in[8];       // [256,300]
  const float*     mlp1_b = (const float*)d_in[9];
  const float*     mlp2_W = (const float*)d_in[10];      // [64,256]
  const float*     mlp2_b = (const float*)d_in[11];
  const float*     out_W  = (const float*)d_in[12];      // [1,64]
  const float*     out_b  = (const float*)d_in[13];

  // ---------------- workspace layout (float offsets) ----------------
  float* ws     = (float*)d_ws;
  float* h      = ws;                  // NN2*LDH = 22,407,168
  float* t      = ws +  22407168;      // NN*DD   = 20,000,000
  float* agg    = ws +  42407168;      // NN2*LDH = 22,407,168
  float* gi     = ws +  64814336;      // NN*600  = 60,000,000
  float* gh     = ws + 124814336;      // NN*600  = 60,000,000
  float* pooled = ws + 184814336;      // NG*64   = 8,192
  // packed f16 weights
  _Float16* fp      = (_Float16*)(ws + 184822528);
  _Float16* Bp_ggc  = fp;                       // 6 * 7*16*512 = 344,064
  _Float16* Bp_wih  = Bp_ggc + 344064;          // 7*40*512    = 143,360
  _Float16* Bp_whh  = Bp_wih + 143360;          // 7*40*512    = 143,360
  _Float16* Bp_mlp1 = Bp_whh + 143360;          // 10*16*512   = 81,920
  _Float16* Bp_mlp2 = Bp_mlp1 + 81920;          // 8*4*512     = 16,384
  // reuse GGC buffers for readout phase
  float* feat  = t;    // NN2*LDF = 32,010,240 (spans t+agg)
  float* feat2 = gi;   // NN2*256 = 25,608,192
  float* feat3 = gh;   // NN*64   =  6,400,000

  const int MT = NN2 / 64;  // 1563 M-tiles

  // ---------------- pack weights into WMMA fragment order ----------------
  {
    long long n;
    n = 6LL * 7 * 16 * 512;  // all 6 GGC layers share one padded shape
    for (int l = 0; l < NLAYER; ++l) {
      long long nl = 7LL * 16 * 512;
      pack_B_kernel<<<(unsigned)((nl + 255) / 256), 256, 0, stream>>>(
          ggc_w + (size_t)l * DD * DD, Bp_ggc + (size_t)l * 7 * 16 * 512,
          DD, DD, DD, 0, 7, 16);
    }
    n = 7LL * 40 * 512;
    pack_B_kernel<<<(unsigned)((n + 255) / 256), 256, 0, stream>>>(
        W_ih, Bp_wih, DD, 3 * DD, DD, 1, 7, 40);
    pack_B_kernel<<<(unsigned)((n + 255) / 256), 256, 0, stream>>>(
        W_hh, Bp_whh, DD, 3 * DD, DD, 1, 7, 40);
    n = 10LL * 16 * 512;
    pack_B_kernel<<<(unsigned)((n + 255) / 256), 256, 0, stream>>>(
        mlp1_W, Bp_mlp1, DD + EF, 256, DD + EF, 1, 10, 16);
    n = 8LL * 4 * 512;
    pack_B_kernel<<<(unsigned)((n + 255) / 256), 256, 0, stream>>>(
        mlp2_W, Bp_mlp2, 256, 64, 256, 1, 8, 4);
  }

  // ---------------- h = pad(x) ----------------
  {
    long long n = (long long)NN2 * LDH;
    init_h_kernel<<<(unsigned)((n + 255) / 256), 256, 0, stream>>>(x, h);
  }

  // ---------------- GGC layers ----------------
  for (int l = 0; l < NLAYER; ++l) {
    // t = h @ W_l   [NN,200]
    wmma_gemm_kernel<0><<<dim3(MT, 4), 256, 0, stream>>>(
        h, Bp_ggc + (size_t)l * 7 * 16 * 512, nullptr, t,
        NN, LDH, DD, LDH, DD, 16);

    // agg = segment_sum(t[src], dst)
    hipMemsetAsync(agg, 0, sizeof(float) * (size_t)NN2 * LDH, stream);
    {
      long long n = (long long)NEDGE * (DD / 4);
      edge_kernel<<<(unsigned)((n + 255) / 256), 256, 0, stream>>>(t, eidx, agg);
    }

    // gi = agg @ W_ih^T + b_ih ; gh = h @ W_hh^T + b_hh   [NN,600]
    wmma_gemm_kernel<0><<<dim3(MT, 10), 256, 0, stream>>>(
        agg, Bp_wih, b_ih, gi, NN, LDH, 3 * DD, LDH, 3 * DD, 40);
    wmma_gemm_kernel<0><<<dim3(MT, 10), 256, 0, stream>>>(
        h, Bp_whh, b_hh, gh, NN, LDH, 3 * DD, LDH, 3 * DD, 40);

    // h = GRU(gi, gh, h)
    {
      long long n = (long long)NN * DD;
      gru_kernel<<<(unsigned)((n + 255) / 256), 256, 0, stream>>>(gi, gh, h);
    }
  }

  // ---------------- readout ----------------
  {
    long long n = (long long)NN2 * LDF;
    concat_kernel<<<(unsigned)((n + 255) / 256), 256, 0, stream>>>(h, x, feat);
  }
  wmma_gemm_kernel<1><<<dim3(MT, 4), 256, 0, stream>>>(
      feat, Bp_mlp1, mlp1_b, feat2, NN2, LDF, 256, LDF, 256, 16);
  wmma_gemm_kernel<1><<<dim3(MT, 1), 256, 0, stream>>>(
      feat2, Bp_mlp2, mlp2_b, feat3, NN, 256, 64, 256, 64, 4);

  hipMemsetAsync(pooled, 0, sizeof(float) * NG * 64, stream);
  {
    long long n = (long long)NN * 64;
    pool_kernel<<<(unsigned)((n + 255) / 256), 256, 0, stream>>>(feat3, batch, pooled);
  }
  final_kernel<<<NG, 64, 0, stream>>>(pooled, out_W, out_b, (float*)d_out);
}
